// NeuralControlActor_39049842655976
// MI455X (gfx1250) — matrix operand order repaired and checked by reading, not compiled, not run on hardware
//
#include <hip/hip_runtime.h>
#include <hip/hip_bf16.h>

// ---- problem constants (from reference) ----
#define T_SEQ     32
#define H_DIM     1024
#define S_DIM     4096
#define D_INNER   2048
#define D_STATE   128
#define D_CONV    4
#define HEADDIM   64
#define NHEADS    32
#define CONV_DIM  (D_INNER + 2 * D_STATE)              // 2304
#define D_IN_PROJ (2 * D_INNER + 2 * D_STATE + NHEADS) // 4384
#define ASIZE_OUT 16384
#define NLAYERS   4
#define KSPLIT    4

typedef __attribute__((ext_vector_type(16))) _Float16 v16h;
typedef __attribute__((ext_vector_type(8)))  float    v8f;

__device__ __forceinline__ float siluf(float v) { return v / (1.f + __expf(-v)); }

__device__ __forceinline__ v16h pack16(float4 x0, float4 x1, float4 x2, float4 x3) {
  v16h r;
  r[0]  = (_Float16)x0.x; r[1]  = (_Float16)x0.y; r[2]  = (_Float16)x0.z; r[3]  = (_Float16)x0.w;
  r[4]  = (_Float16)x1.x; r[5]  = (_Float16)x1.y; r[6]  = (_Float16)x1.z; r[7]  = (_Float16)x1.w;
  r[8]  = (_Float16)x2.x; r[9]  = (_Float16)x2.y; r[10] = (_Float16)x2.z; r[11] = (_Float16)x2.w;
  r[12] = (_Float16)x3.x; r[13] = (_Float16)x3.y; r[14] = (_Float16)x3.z; r[15] = (_Float16)x3.w;
  return r;
}

// ---------------------------------------------------------------------------
// GEMV: out[n] = act( dot(W[n,:], x) + b[n] ), one block per output row.
// act: 0 = none, 1 = relu, 2 = sigmoid
// ---------------------------------------------------------------------------
__global__ void gemv_rows(const float* __restrict__ W, const float* __restrict__ x,
                          const float* __restrict__ b, float* __restrict__ out,
                          int K, int act) {
  __shared__ float red[256];
  const int n   = blockIdx.x;
  const int tid = threadIdx.x;
  const float4* Wr = (const float4*)(W + (size_t)n * K);
  const float4* xv = (const float4*)x;
  const int K4 = K >> 2;
  float s = 0.f;
  for (int k = tid; k < K4; k += 256) {
    if (k + 512 < K4) __builtin_prefetch(&Wr[k + 512], 0, 0); // global_prefetch_b8
    float4 w4 = Wr[k];
    float4 v4 = xv[k];
    s += w4.x * v4.x + w4.y * v4.y + w4.z * v4.z + w4.w * v4.w;
  }
  red[tid] = s;
  __syncthreads();
  for (int off = 128; off > 0; off >>= 1) {
    if (tid < off) red[tid] += red[tid + off];
    __syncthreads();
  }
  if (tid == 0) {
    float v = red[0] + (b ? b[n] : 0.f);
    if (act == 1)      v = fmaxf(v, 0.f);
    else if (act == 2) v = 1.f / (1.f + __expf(-v));
    out[n] = v;
  }
}

// ---------------------------------------------------------------------------
// K-split WMMA GEMM partial: Part[slice][M, n0+strip] = A[M, kslice] * W^T.
// blockIdx.y = K slice (KSPLIT slices of kcLen each). Launches are wave-exact:
// no guards, EXEC all-1s for every WMMA.
// Loads are issued as ONE inline-asm s_clause of 4+4*NT global_load_b128 into
// forced-distinct staging registers, with a single s_wait_loadcnt per K-step
// -> 20 loads in flight per wave instead of 1.
// Fragment layouts: 16-bit A/B use k = 16*half + s in both (not the ISA table order); f32 C/D per ISA 7.12.2.
// ---------------------------------------------------------------------------
template <int NT>
__global__ void __launch_bounds__(128, 1)
wmma_gemm_part(const float* __restrict__ A, const float* __restrict__ W,
               float* __restrict__ Part, int M, int N, int K, int n0, int kcLen) {
  const int wavesPerBlock = blockDim.x >> 5;
  const int waveId = blockIdx.x * wavesPerBlock + ((int)threadIdx.x >> 5);
  const int tilesM = M >> 4;
  const int tm   = waveId % tilesM;
  const int gn   = waveId / tilesM;
  const int col0 = n0 + gn * (NT * 16);
  const int lane = threadIdx.x & 31;
  const int idx  = lane & 15;   // M row (A) / N col (B,C,D)
  const int kh   = lane >> 4;   // lane-half selects K sub-block / M sub-block of C
  const int kBase = blockIdx.y * kcLen;
  float* Out = Part + (size_t)blockIdx.y * M * N;

  // A 16x32: lane-half 0 holds K=0..15; half 1 holds K=16..31
  const float* pa  = A + (size_t)(tm * 16 + idx) * K + kBase + kh * 16;
  // B 32x16: lane-half 0 holds K=0..15; half 1 holds K=16..31 (col = idx)
  const float* pb0 = W + (size_t)(col0 + idx) * K + kBase + kh * 16;
  const float* pb1 = pb0 + (size_t)16 * K;
  const float* pb2 = pb0 + (size_t)32 * K;
  const float* pb3 = pb0 + (size_t)48 * K;

  v8f c[NT];
#pragma unroll
  for (int nt = 0; nt < NT; ++nt) c[nt] = (v8f){};

  float4 ar[4];
  float4 br[4][4];

  // issue all staged loads for one K-step as a single clause (no waits)
  auto issue = [&]() {
    if (NT == 4) {
      asm volatile(
        "s_clause 0x13\n\t"
        "global_load_b128 %0, %20, off\n\t"
        "global_load_b128 %1, %20, off offset:16\n\t"
        "global_load_b128 %2, %20, off offset:32\n\t"
        "global_load_b128 %3, %20, off offset:48\n\t"
        "global_load_b128 %4, %21, off\n\t"
        "global_load_b128 %5, %21, off offset:16\n\t"
        "global_load_b128 %6, %21, off offset:32\n\t"
        "global_load_b128 %7, %21, off offset:48\n\t"
        "global_load_b128 %8, %22, off\n\t"
        "global_load_b128 %9, %22, off offset:16\n\t"
        "global_load_b128 %10, %22, off offset:32\n\t"
        "global_load_b128 %11, %22, off offset:48\n\t"
        "global_load_b128 %12, %23, off\n\t"
        "global_load_b128 %13, %23, off offset:16\n\t"
        "global_load_b128 %14, %23, off offset:32\n\t"
        "global_load_b128 %15, %23, off offset:48\n\t"
        "global_load_b128 %16, %24, off\n\t"
        "global_load_b128 %17, %24, off offset:16\n\t"
        "global_load_b128 %18, %24, off offset:32\n\t"
        "global_load_b128 %19, %24, off offset:48"
        : "=v"(ar[0]), "=v"(ar[1]), "=v"(ar[2]), "=v"(ar[3]),
          "=v"(br[0][0]), "=v"(br[0][1]), "=v"(br[0][2]), "=v"(br[0][3]),
          "=v"(br[1][0]), "=v"(br[1][1]), "=v"(br[1][2]), "=v"(br[1][3]),
          "=v"(br[2][0]), "=v"(br[2][1]), "=v"(br[2][2]), "=v"(br[2][3]),
          "=v"(br[3][0]), "=v"(br[3][1]), "=v"(br[3][2]), "=v"(br[3][3])
        : "v"(pa), "v"(pb0), "v"(pb1), "v"(pb2), "v"(pb3)
        : "memory");
    } else { // NT == 2
      asm volatile(
        "s_clause 0xB\n\t"
        "global_load_b128 %0, %12, off\n\t"
        "global_load_b128 %1, %12, off offset:16\n\t"
        "global_load_b128 %2, %12, off offset:32\n\t"
        "global_load_b128 %3, %12, off offset:48\n\t"
        "global_load_b128 %4, %13, off\n\t"
        "global_load_b128 %5, %13, off offset:16\n\t"
        "global_load_b128 %6, %13, off offset:32\n\t"
        "global_load_b128 %7, %13, off offset:48\n\t"
        "global_load_b128 %8, %14, off\n\t"
        "global_load_b128 %9, %14, off offset:16\n\t"
        "global_load_b128 %10, %14, off offset:32\n\t"
        "global_load_b128 %11, %14, off offset:48"
        : "=v"(ar[0]), "=v"(ar[1]), "=v"(ar[2]), "=v"(ar[3]),
          "=v"(br[0][0]), "=v"(br[0][1]), "=v"(br[0][2]), "=v"(br[0][3]),
          "=v"(br[1][0]), "=v"(br[1][1]), "=v"(br[1][2]), "=v"(br[1][3])
        : "v"(pa), "v"(pb0), "v"(pb1)
        : "memory");
    }
  };

  issue();
  pa += 32; pb0 += 32; pb1 += 32; pb2 += 32; pb3 += 32;

  for (int k0 = 0; k0 < kcLen; k0 += 32) {
    // single wait per K-step; volatile asm is a scheduling boundary between
    // the load clause (before) and the fragment packs (after)
    asm volatile("s_wait_loadcnt 0x0" ::: "memory");
    const v16h a = pack16(ar[0], ar[1], ar[2], ar[3]);
#pragma unroll
    for (int nt = 0; nt < NT; ++nt) {
      const v16h b = pack16(br[nt][0], br[nt][1], br[nt][2], br[nt][3]);
      c[nt] = __builtin_amdgcn_wmma_f32_16x16x32_f16(false, a, false, b, (short)0,
                                                     c[nt], false, false);
    }
    if (k0 + 32 < kcLen) {
      issue();  // next K-step in flight while the XDL WMMAs drain
      pa += 32; pb0 += 32; pb1 += 32; pb2 += 32; pb3 += 32;
    }
  }

#pragma unroll
  for (int nt = 0; nt < NT; ++nt) {
    const int n = col0 + nt * 16 + idx;
#pragma unroll
    for (int r = 0; r < 8; ++r) {
      const int m = tm * 16 + kh * 8 + r; // C/D: lanes 0-15 -> M=r, lanes 16-31 -> M=8+r
      Out[(size_t)m * N + n] = c[nt][r];
    }
  }
}

// out[i] = sum_s part[s][i]  (fixed order -> deterministic)
__global__ void ksplit_reduce(const float* __restrict__ part, float* __restrict__ out,
                              int MN) {
  const int i = blockIdx.x * blockDim.x + threadIdx.x;
  if (i < MN) {
    float s = 0.f;
#pragma unroll
    for (int j = 0; j < KSPLIT; ++j) s += part[(size_t)j * MN + i];
    out[i] = s;
  }
}

// ---------------------------------------------------------------------------
// Depthwise causal conv (D_CONV=4) over xBC columns of zxbcdt, + bias + SiLU.
// ---------------------------------------------------------------------------
__global__ void conv_silu(const float* __restrict__ zxbcdt, const float* __restrict__ convw,
                          const float* __restrict__ convb, float* __restrict__ xBC) {
  const int idx = blockIdx.x * blockDim.x + threadIdx.x;
  if (idx >= T_SEQ * CONV_DIM) return;
  const int c = idx % CONV_DIM;
  const int t = idx / CONV_DIM;
  float acc = convb[c];
#pragma unroll
  for (int k = 0; k < D_CONV; ++k) {
    const int ts = t + k - (D_CONV - 1);
    if (ts >= 0)
      acc += zxbcdt[(size_t)ts * D_IN_PROJ + D_INNER + c] * convw[c * D_CONV + k];
  }
  xBC[idx] = siluf(acc);
}

// ---------------------------------------------------------------------------
// Selective-state scan. One block (256 threads) per head.
// ---------------------------------------------------------------------------
__global__ void ssm_scan(const float* __restrict__ xBC, const float* __restrict__ zxbcdt,
                         const float* __restrict__ dtbias, const float* __restrict__ Alog,
                         const float* __restrict__ Dp, float* __restrict__ y) {
  __shared__ float Bs[D_STATE], Cs[D_STATE], yp[4 * HEADDIM];
  const int head = blockIdx.x;
  const int tid  = threadIdx.x;
  const int p    = tid & 63;
  const int g    = tid >> 6;
  const float A   = -__expf(Alog[head]);
  const float dtb = dtbias[head];
  const float Dh  = Dp[head];
  float h[32];
#pragma unroll
  for (int j = 0; j < 32; ++j) h[j] = 0.f;

  for (int t = 0; t < T_SEQ; ++t) {
    if (tid < D_STATE)          Bs[tid]           = xBC[(size_t)t * CONV_DIM + D_INNER + tid];
    else if (tid < 2 * D_STATE) Cs[tid - D_STATE] = xBC[(size_t)t * CONV_DIM + D_INNER + tid];
    __syncthreads();
    float dtraw = zxbcdt[(size_t)t * D_IN_PROJ + (D_IN_PROJ - NHEADS) + head] + dtb;
    float dt    = (dtraw > 20.f) ? dtraw : __logf(1.f + __expf(dtraw)); // softplus
    float dA    = __expf(dt * A);
    float xv    = xBC[(size_t)t * CONV_DIM + head * HEADDIM + p];
    float dtx   = dt * xv;
    float acc   = 0.f;
#pragma unroll
    for (int j = 0; j < 32; ++j) {
      const int n = g * 32 + j;
      h[j] = h[j] * dA + dtx * Bs[n];
      acc += h[j] * Cs[n];
    }
    yp[g * HEADDIM + p] = acc;
    __syncthreads();
    if (g == 0) {
      float yt = yp[p] + yp[64 + p] + yp[128 + p] + yp[192 + p] + Dh * xv;
      y[(size_t)t * D_INNER + head * HEADDIM + p] = yt;
    }
    __syncthreads();
  }
}

// ---------------------------------------------------------------------------
// yg = y * silu(z); RMS-normalize over D_INNER; scale by normw. One block/row.
// ---------------------------------------------------------------------------
__global__ void gate_rmsnorm(const float* __restrict__ y, const float* __restrict__ zxbcdt,
                             const float* __restrict__ normw, float* __restrict__ yn) {
  __shared__ float red[256];
  const int t   = blockIdx.x;
  const int tid = threadIdx.x;
  float ss = 0.f;
  for (int c = tid; c < D_INNER; c += 256) {
    float z = zxbcdt[(size_t)t * D_IN_PROJ + c];
    float v = y[(size_t)t * D_INNER + c] * siluf(z);
    yn[(size_t)t * D_INNER + c] = v;
    ss += v * v;
  }
  red[tid] = ss;
  __syncthreads();
  for (int off = 128; off > 0; off >>= 1) {
    if (tid < off) red[tid] += red[tid + off];
    __syncthreads();
  }
  const float scale = rsqrtf(red[0] / (float)D_INNER + 1e-5f);
  for (int c = tid; c < D_INNER; c += 256) {
    yn[(size_t)t * D_INNER + c] *= scale * normw[c];
  }
}

// ---------------------------------------------------------------------------
extern "C" void kernel_launch(void* const* d_in, const int* in_sizes, int n_in,
                              void* d_out, int out_size, void* d_ws, size_t ws_size,
                              hipStream_t stream) {
  const float* x        = (const float*)d_in[0];
  const float* w1       = (const float*)d_in[1];
  const float* b1       = (const float*)d_in[2];
  const float* w2       = (const float*)d_in[3];
  const float* b2       = (const float*)d_in[4];
  const float* m_Win    = (const float*)d_in[5];
  const float* m_convw  = (const float*)d_in[6];
  const float* m_convb  = (const float*)d_in[7];
  const float* m_dtbias = (const float*)d_in[8];
  const float* m_Alog   = (const float*)d_in[9];
  const float* m_D      = (const float*)d_in[10];
  const float* m_normw  = (const float*)d_in[11];
  const float* m_Wout   = (const float*)d_in[12];
  const float* w3       = (const float*)d_in[13];
  const float* b3       = (const float*)d_in[14];
  const float* w4       = (const float*)d_in[15];
  const float* b4       = (const float*)d_in[16];
  float* out = (float*)d_out;

  float* ws = (float*)d_ws;
  float* buf_h    = ws; ws += H_DIM;
  float* buf_seq  = ws; ws += T_SEQ * H_DIM;
  float* buf_zx   = ws; ws += T_SEQ * D_IN_PROJ;
  float* buf_xBC  = ws; ws += T_SEQ * CONV_DIM;
  float* buf_y    = ws; ws += T_SEQ * D_INNER;
  float* buf_yn   = ws; ws += T_SEQ * D_INNER;
  float* buf_fh   = ws; ws += H_DIM;
  float* buf_part = ws; ws += KSPLIT * T_SEQ * D_IN_PROJ; // covers both GEMMs

  // h = relu(x @ w1.T + b1)
  gemv_rows<<<H_DIM, 256, 0, stream>>>(w1, x, b1, buf_h, S_DIM, 1);
  // seq = (h @ w2.T + b2).reshape(T, H)
  gemv_rows<<<T_SEQ * H_DIM, 256, 0, stream>>>(w2, buf_h, b2, buf_seq, H_DIM, 0);

  for (int l = 0; l < NLAYERS; ++l) {
    const float* Win   = m_Win   + (size_t)l * D_IN_PROJ * H_DIM;
    const float* convw = m_convw + (size_t)l * CONV_DIM * D_CONV;
    const float* convb = m_convb + (size_t)l * CONV_DIM;
    const float* dtb   = m_dtbias+ (size_t)l * NHEADS;
    const float* Alog  = m_Alog  + (size_t)l * NHEADS;
    const float* Dp    = m_D     + (size_t)l * NHEADS;
    const float* normw = m_normw + (size_t)l * D_INNER;
    const float* Wout  = m_Wout  + (size_t)l * H_DIM * D_INNER;

    // zxbcdt = seq @ Win.T  (M=32, N=4384, K=1024) — WMMA, K-split x4
    {
      // main: 272 of 274 N-tiles; waves/slice = 2*68 = 136 -> 34 blocks x 128
      wmma_gemm_part<4><<<dim3(34, KSPLIT), 128, 0, stream>>>(
          buf_seq, Win, buf_part, T_SEQ, D_IN_PROJ, H_DIM, 0, H_DIM / KSPLIT);
      // tail: last 2 N-tiles; waves/slice = 2 -> 1 block x 64
      wmma_gemm_part<2><<<dim3(1, KSPLIT), 64, 0, stream>>>(
          buf_seq, Win, buf_part, T_SEQ, D_IN_PROJ, H_DIM, 68 * 64, H_DIM / KSPLIT);
      const int MN = T_SEQ * D_IN_PROJ;
      ksplit_reduce<<<(MN + 255) / 256, 256, 0, stream>>>(buf_part, buf_zx, MN);
    }
    conv_silu<<<(T_SEQ * CONV_DIM + 255) / 256, 256, 0, stream>>>(buf_zx, convw, convb, buf_xBC);
    ssm_scan<<<NHEADS, 256, 0, stream>>>(buf_xBC, buf_zx, dtb, Alog, Dp, buf_y);
    gate_rmsnorm<<<T_SEQ, 256, 0, stream>>>(buf_y, buf_zx, normw, buf_yn);
    // seq = yn @ Wout.T  (M=32, N=1024, K=2048) — WMMA, K-split x4
    {
      wmma_gemm_part<4><<<dim3(8, KSPLIT), 128, 0, stream>>>(
          buf_yn, Wout, buf_part, T_SEQ, H_DIM, D_INNER, 0, D_INNER / KSPLIT);
      const int MN = T_SEQ * H_DIM;
      ksplit_reduce<<<(MN + 255) / 256, 256, 0, stream>>>(buf_part, buf_seq, MN);
    }
  }

  // hidden = relu(flat @ w3.T + b3); out = sigmoid(hidden @ w4.T + b4)
  gemv_rows<<<H_DIM, 256, 0, stream>>>(w3, buf_seq, b3, buf_fh, T_SEQ * H_DIM, 1);
  gemv_rows<<<ASIZE_OUT, 256, 0, stream>>>(w4, buf_fh, b4, out, H_DIM, 2);
}
